// Block_88656714925492
// MI455X (gfx1250) — matrix operand order repaired and checked by reading, not compiled, not run on hardware
//
#include <hip/hip_runtime.h>
#include <hip/hip_bf16.h>
#include <math.h>

typedef unsigned short u16;
typedef unsigned int   u32;
typedef __attribute__((ext_vector_type(16))) __bf16 bf16x16;
typedef __attribute__((ext_vector_type(8)))  float  f32x8;

#define CB   256   // channels
#define FDC  1024  // ffn dim
#define BNB  32    // batch
#define NHH  8     // heads

// ---------------- helpers ----------------
__device__ __forceinline__ u16 f2bf(float f) {
  u32 u = __float_as_uint(f);
  u32 r = u + 0x7FFFu + ((u >> 16) & 1u);
  return (u16)(r >> 16);
}
__device__ __forceinline__ float bf2f(u16 h) {
  return __uint_as_float(((u32)h) << 16);
}
__device__ __forceinline__ float gelu_f(float x) {
  return 0.5f * x * (1.0f + erff(x * 0.7071067811865476f));
}
__device__ __forceinline__ float block_reduce_sum(float v, float* red) {
  int c = threadIdx.x;
  red[c] = v; __syncthreads();
  for (int st = 128; st > 0; st >>= 1) {
    if (c < st) red[c] += red[c + st];
    __syncthreads();
  }
  float r = red[0];
  __syncthreads();
  return r;
}

// ---------------- WMMA GEMM: out = epi(A[M,K](bf16) @ W[N,K]^T + bias) ----------------
// 128x128 block tile, 8 waves, each wave owns 32(M)x64(N) = 8 WMMA per K-step.
// Double-buffered LDS (1 barrier per K-step); branchless clamped tail so the
// steady-state loop is straight-line: sync -> prefetch -> tile store -> frags -> 8 WMMA.
#define BM 128
#define BN 128
#define BK 32

union FragB16 { uint4 q[2]; bf16x16 v; };

template<int EPI> // 0 = bias, 1 = bias + exact GELU
__global__ __launch_bounds__(256) void gemm_bf16_wmma(
    const u16* __restrict__ A, const u16* __restrict__ W,
    const float* __restrict__ bias,
    float* __restrict__ outF, u16* __restrict__ outH,
    int M, int N, int K)
{
  __shared__ __align__(16) u16 As[2][BM * BK];
  __shared__ __align__(16) u16 Bs[2][BN * BK];

  const int tidx = threadIdx.x;
  const int lane = tidx & 31;
  const int wave = tidx >> 5;    // 0..7
  const int wm   = wave >> 1;    // 0..3 -> 32 output rows
  const int wn   = wave & 1;     // 0..1 -> 64 output cols
  const int tileM = blockIdx.x * BM;
  const int tileN = blockIdx.y * BN;

  // cooperative tile loaders: 128 rows x 32 bf16 = 512 uint4; 2 per thread
  const int lrow = tidx >> 1;          // 0..127
  const int lcb  = (tidx & 1) * 2;     // uint4 index 0 or 2 within row
  const u16* gA = A + (size_t)(tileM + lrow) * K + lcb * 8;
  const u16* gB = W + (size_t)(tileN + lrow) * K + lcb * 8;

  f32x8 acc[2][4];
  for (int i = 0; i < 2; i++)
    for (int j = 0; j < 4; j++)
      for (int r = 0; r < 8; r++) acc[i][j][r] = 0.0f;

  auto load_tile = [&](int k0, int buf) {
    const uint4* sa = (const uint4*)(gA + k0);
    const uint4* sb = (const uint4*)(gB + k0);
    uint4* da = ((uint4*)As[buf]) + lrow * 4 + lcb;
    uint4* db = ((uint4*)Bs[buf]) + lrow * 4 + lcb;
    da[0] = sa[0]; da[1] = sa[1];
    db[0] = sb[0]; db[1] = sb[1];
  };

  const int KT = K / BK;
  load_tile(0, 0);
  int cur = 0;
  for (int kt = 0; kt < KT; kt++) {
    __syncthreads();                          // buf[cur] ready for everyone

    // branchless clamped prefetch (K+2 tile) and next-tile fill (K+1 tile).
    // Final iterations re-touch the last tile into a never-read buffer: harmless.
    int kpf = (kt + 2 < KT ? kt + 2 : KT - 1) * BK;
    __builtin_prefetch(gA + kpf, 0, 1);
    __builtin_prefetch(gB + kpf, 0, 1);
    int knx = (kt + 1 < KT ? kt + 1 : KT - 1) * BK;
    load_tile(knx, cur ^ 1);

    FragB16 fa[2], fb[4];
    {
      const int kbA = (lane >> 4) * 8;   // A: lanes 0-15 K{0..7,16..23}, 16-31 K{8..15,24..31}
      for (int ms = 0; ms < 2; ms++) {
        const u16* p = &As[cur][(wm * 32 + ms * 16 + (lane & 15)) * BK];
        fa[ms].q[0] = *(const uint4*)(p + kbA);
        fa[ms].q[1] = *(const uint4*)(p + kbA + 16);
      }
      const int kbB = (lane >> 4) * 8;   // B: lanes 0-15 K{0..7,16..23}, 16-31 K{8..15,24..31}
      for (int ns = 0; ns < 4; ns++) {
        const u16* p = &Bs[cur][(wn * 64 + ns * 16 + (lane & 15)) * BK];
        fb[ns].q[0] = *(const uint4*)(p + kbB);
        fb[ns].q[1] = *(const uint4*)(p + kbB + 16);
      }
    }
    for (int ms = 0; ms < 2; ms++)
      for (int ns = 0; ns < 4; ns++)
        acc[ms][ns] = __builtin_amdgcn_wmma_f32_16x16x32_bf16(
            false, fa[ms].v, false, fb[ns].v, (short)0, acc[ms][ns], false, false);
    cur ^= 1;
  }

  // epilogue: C/D layout -> VGPR r: M = r (+8 for lanes>=16), N = lane&15
  for (int ms = 0; ms < 2; ms++)
    for (int ns = 0; ns < 4; ns++) {
      int col = tileN + wn * 64 + ns * 16 + (lane & 15);
      float bv = bias[col];
      for (int r = 0; r < 8; r++) {
        int row = tileM + wm * 32 + ms * 16 + r + ((lane >> 4) << 3);
        float v = acc[ms][ns][r] + bv;
        if (EPI == 1) v = gelu_f(v);
        size_t o = (size_t)row * N + col;
        if (outF) outF[o] = v;
        if (outH) outH[o] = f2bf(v);
      }
    }
}

// ---------------- NCHW input + positional dwconv3 -> token-major fp32 ----------------
__global__ __launch_bounds__(256) void posadd_kernel(
    const float* __restrict__ in, const float* __restrict__ w, const float* __restrict__ bias,
    float* __restrict__ out_tm, int H, int W)
{
  size_t t = (size_t)blockIdx.x * blockDim.x + threadIdx.x;
  size_t total = (size_t)BNB * H * W * CB;
  if (t >= total) return;
  int c = (int)(t % CB);
  size_t m = t / CB;
  int Np = H * W;
  int p = (int)(m % Np);
  int b = (int)(m / Np);
  int hh = p / W, ww = p % W;
  const float* plane = in + ((size_t)b * CB + c) * Np;
  float acc = plane[p] + bias[c];
  for (int dh = -1; dh <= 1; dh++)
    for (int dw = -1; dw <= 1; dw++) {
      int y = hh + dh, x = ww + dw;
      if (y >= 0 && y < H && x >= 0 && x < W)
        acc += plane[y * W + x] * w[c * 9 + (dh + 1) * 3 + (dw + 1)];
    }
  out_tm[m * CB + c] = acc;
}

// ---------------- row LayerNorm over C=256 (one block per token) ----------------
template<bool F32OUT, bool HOUT, bool SCORES>
__global__ __launch_bounds__(256) void ln_rows(
    const float* __restrict__ in, const float* __restrict__ g, const float* __restrict__ bb,
    float* __restrict__ outF, u16* __restrict__ outH, float* __restrict__ scores)
{
  __shared__ float red[256];
  size_t m = blockIdx.x;
  int c = threadIdx.x;
  float v = in[m * CB + c];
  float mu = block_reduce_sum(v, red) * (1.0f / CB);
  float d = v - mu;
  float var = block_reduce_sum(d * d, red) * (1.0f / CB);
  float y = d * rsqrtf(var + 1e-6f) * g[c] + bb[c];
  if (F32OUT) outF[m * CB + c] = y;
  if (HOUT)  outH[m * CB + c] = f2bf(y);
  if (SCORES) {
    float ss = block_reduce_sum(y * y, red);
    if (c == 0) scores[m] = ss;
  }
}

// ---------------- per-batch channel sums ----------------
__global__ __launch_bounds__(256) void colsum_kernel(
    const float* __restrict__ ln, float* __restrict__ chansum, int Np)
{
  int b = blockIdx.x, c = threadIdx.x;
  float s = 0.f;
  const float* p = ln + (size_t)b * Np * CB + c;
  for (int i = 0; i < Np; i++) s += p[(size_t)i * CB];
  chansum[b * CB + c] = s;
}

// ---------------- per-batch iterative top-k ----------------
__global__ __launch_bounds__(256) void topk_kernel(
    const float* __restrict__ scores, int* __restrict__ idx, int Np, int k)
{
  __shared__ float s[1024];
  __shared__ float rv[256];
  __shared__ int   ri[256];
  int b = blockIdx.x, t = threadIdx.x;
  for (int i = t; i < Np; i += 256) s[i] = scores[(size_t)b * Np + i];
  __syncthreads();
  for (int it = 0; it < k; it++) {
    float best = -3.4e38f; int bi = 0x7fffffff;
    for (int i = t; i < Np; i += 256) {
      float v = s[i];
      if (v > best || (v == best && i < bi)) { best = v; bi = i; }
    }
    rv[t] = best; ri[t] = bi; __syncthreads();
    for (int st = 128; st > 0; st >>= 1) {
      if (t < st) {
        if (rv[t + st] > rv[t] || (rv[t + st] == rv[t] && ri[t + st] < ri[t])) {
          rv[t] = rv[t + st]; ri[t] = ri[t + st];
        }
      }
      __syncthreads();
    }
    if (t == 0) { idx[b * k + it] = ri[0]; s[ri[0]] = -3.4e38f; }
    __syncthreads();
  }
}

// ---------------- token gather + background mean, pack to bf16 ----------------
__global__ __launch_bounds__(256) void gather_tokens(
    const float* __restrict__ ln, const float* __restrict__ chansum, const int* __restrict__ idx,
    u16* __restrict__ tok, int Np, int k, int Npad)
{
  int b = blockIdx.x, c = threadIdx.x;
  float sel = 0.f;
  for (int i = 0; i < k; i++) {
    int p = idx[b * k + i];
    float v = ln[((size_t)b * Np + p) * CB + c];
    tok[((size_t)(b * Npad + i)) * CB + c] = f2bf(v);
    sel += v;
  }
  float bg = (chansum[b * CB + c] - sel) / (float)(Np - k);
  tok[((size_t)(b * Npad + k)) * CB + c] = f2bf(bg);
  for (int i = k + 1; i < Npad; i++) tok[((size_t)(b * Npad + i)) * CB + c] = 0;
}

// ---------------- small attention: one block per (batch, head), 2-pass softmax ----------------
#define NMAX 112
__global__ __launch_bounds__(128) void attention_kernel(
    const float* __restrict__ qkv, u16* __restrict__ o_tok, int N, int Npad)
{
  __shared__ float qs[NMAX * 32];
  __shared__ float ks[NMAX * 32];
  __shared__ float vs[NMAX * 32];
  int b = blockIdx.x, h = blockIdx.y;
  const float scale = 0.17677669529663687f; // 1/sqrt(32)
  for (int i = threadIdx.x; i < N * 32; i += 128) {
    int n = i >> 5, d = i & 31;
    size_t row = (size_t)(b * Npad + n) * (3 * CB);
    qs[i] = qkv[row + h * 32 + d];
    ks[i] = qkv[row + CB + h * 32 + d];
    vs[i] = qkv[row + 2 * CB + h * 32 + d];
  }
  __syncthreads();
  int n = threadIdx.x;
  if (n < N) {
    float mx = -3.4e38f;
    for (int m = 0; m < N; m++) {
      float s = 0.f;
      for (int d = 0; d < 32; d++) s += qs[n * 32 + d] * ks[m * 32 + d];
      mx = fmaxf(mx, s * scale);
    }
    float Z = 0.f, acc[32];
    for (int d = 0; d < 32; d++) acc[d] = 0.f;
    for (int m = 0; m < N; m++) {
      float s = 0.f;
      for (int d = 0; d < 32; d++) s += qs[n * 32 + d] * ks[m * 32 + d];
      float e = __expf(s * scale - mx);
      Z += e;
      for (int d = 0; d < 32; d++) acc[d] += e * vs[m * 32 + d];
    }
    float inv = 1.f / Z;
    for (int d = 0; d < 32; d++)
      o_tok[(size_t)(b * Npad + n) * CB + h * 32 + d] = f2bf(acc[d] * inv);
  }
  // deterministically zero padded token rows
  for (int i = N * 32 + (int)threadIdx.x; i < Npad * 32; i += 128) {
    int nn = i >> 5, d = i & 31;
    o_tok[(size_t)(b * Npad + nn) * CB + h * 32 + d] = 0;
  }
}

// ---------------- batchnorm-eval pack (fp32 ln -> bf16) ----------------
__global__ __launch_bounds__(256) void bnpack_kernel(
    const float* __restrict__ ln, const float* __restrict__ g, const float* __restrict__ bb,
    const float* __restrict__ mean, const float* __restrict__ var,
    u16* __restrict__ out, size_t total, int Cc)
{
  size_t t = (size_t)blockIdx.x * blockDim.x + threadIdx.x;
  if (t >= total) return;
  int c = (int)(t % Cc);
  float v = ln[t];
  out[t] = f2bf((v - mean[c]) * rsqrtf(var[c] + 1e-5f) * g[c] + bb[c]);
}

// ---------------- depthwise 3x3 on token-major bf16 (+optional residual), bf16 out ----------------
__global__ __launch_bounds__(256) void dwconv_tm_kernel(
    const u16* __restrict__ in, const float* __restrict__ w, const float* __restrict__ bias,
    u16* __restrict__ out, int Cc, int H, int W, int addres)
{
  size_t t = (size_t)blockIdx.x * blockDim.x + threadIdx.x;
  size_t total = (size_t)BNB * H * W * Cc;
  if (t >= total) return;
  int c = (int)(t % Cc);
  size_t m = t / Cc;
  int Np = H * W;
  int p = (int)(m % Np);
  int b = (int)(m / Np);
  int hh = p / W, ww = p % W;
  const u16* base = in + (size_t)b * Np * Cc;
  float acc = bias[c];
  for (int dh = -1; dh <= 1; dh++)
    for (int dw = -1; dw <= 1; dw++) {
      int y = hh + dh, x = ww + dw;
      if (y >= 0 && y < H && x >= 0 && x < W)
        acc += bf2f(base[(size_t)(y * W + x) * Cc + c]) * w[c * 9 + (dh + 1) * 3 + (dw + 1)];
    }
  if (addres) acc += bf2f(in[m * Cc + c]);
  out[m * Cc + c] = f2bf(acc);
}

// ---------------- restored + y composition ----------------
__global__ __launch_bounds__(256) void combine_kernel(
    const float* __restrict__ y3, const float* __restrict__ oproj,
    float* __restrict__ comb, int Np, int k, int Npad, size_t total)
{
  size_t t = (size_t)blockIdx.x * blockDim.x + threadIdx.x;
  if (t >= total) return;
  int c = (int)(t % CB);
  size_t m = t / CB;
  int b = (int)(m / Np);
  comb[t] = y3[t] + oproj[((size_t)(b * Npad + k)) * CB + c];
}

__global__ __launch_bounds__(256) void scatter_kernel(
    const float* __restrict__ y3, const float* __restrict__ oproj, const int* __restrict__ idx,
    float* __restrict__ comb, int Np, int k, int Npad)
{
  size_t t = (size_t)blockIdx.x * blockDim.x + threadIdx.x;
  size_t total = (size_t)BNB * k * CB;
  if (t >= total) return;
  int c = (int)(t % CB);
  int i = (int)((t / CB) % k);
  int b = (int)(t / ((size_t)CB * k));
  int p = idx[b * k + i];
  size_t m = (size_t)b * Np + p;
  comb[m * CB + c] = y3[m * CB + c] + oproj[((size_t)(b * Npad + i)) * CB + c];
}

// ---------------- generic glue ----------------
__global__ __launch_bounds__(256) void pack_bf16_kernel(
    const float* __restrict__ src, u16* __restrict__ dst, size_t n)
{
  size_t t = (size_t)blockIdx.x * blockDim.x + threadIdx.x;
  if (t < n) dst[t] = f2bf(src[t]);
}

__global__ __launch_bounds__(256) void add_kernel(
    const float* __restrict__ a, const float* __restrict__ b, float* __restrict__ o, size_t n)
{
  size_t t = (size_t)blockIdx.x * blockDim.x + threadIdx.x;
  if (t < n) o[t] = a[t] + b[t];
}

__global__ __launch_bounds__(256) void final_kernel(
    const float* __restrict__ t2, const float* __restrict__ ffn,
    float* __restrict__ out_nchw, int Np, size_t total)
{
  size_t t = (size_t)blockIdx.x * blockDim.x + threadIdx.x;
  if (t >= total) return;
  int c = (int)(t % CB);
  size_t m = t / CB;
  int p = (int)(m % Np);
  int b = (int)(m / Np);
  out_nchw[((size_t)b * CB + c) * Np + p] = t2[t] + ffn[t];
}

// ================= host driver =================
struct Params {
  const float *pos_w, *pos_b, *qkv_b, *po_b, *bn_g, *bn_b, *bn_m, *bn_v;
  const float *fc1_b, *conv_w, *conv_b, *fc2_b, *proj_b, *f1_b, *fdw_w, *fdw_b, *f2_b;
  const u16 *qkv_wh, *po_wh, *fc1_wh, *fc2_wh, *proj_wh, *f1_wh, *f2_wh;
  float *tA, *tLn, *tY3, *tComb, *tT2;
  u16 *h1, *h2b, *hF1, *hF2, *tok, *otok;
  float *qkvbuf, *oproj, *scores, *chansum;
  int *idx;
};

static void run_tensor(const float* inp, float* outp, int H, int W, int k, int Npad,
                       const float* ln1g, const float* ln1b,
                       const float* ln2g, const float* ln2b,
                       const Params& P, hipStream_t s)
{
  const int Np = H * W;
  const int M = BNB * Np;          // token count (mult of 128)
  const int Mt = BNB * Npad;       // padded token-GEMM rows (mult of 128)
  const size_t MC = (size_t)M * CB;
  const size_t MF = (size_t)M * FDC;
  dim3 b256(256), b128(128);

  posadd_kernel<<<dim3((u32)(MC / 256)), b256, 0, s>>>(inp, P.pos_w, P.pos_b, P.tA, H, W);
  ln_rows<true, false, true><<<dim3((u32)M), b256, 0, s>>>(P.tA, ln1g, ln1b, P.tLn, nullptr, P.scores);
  colsum_kernel<<<dim3(BNB), b256, 0, s>>>(P.tLn, P.chansum, Np);
  topk_kernel<<<dim3(BNB), b256, 0, s>>>(P.scores, P.idx, Np, k);
  gather_tokens<<<dim3(BNB), b256, 0, s>>>(P.tLn, P.chansum, P.idx, P.tok, Np, k, Npad);

  gemm_bf16_wmma<0><<<dim3(Mt / BM, (3 * CB) / BN), b256, 0, s>>>(
      P.tok, P.qkv_wh, P.qkv_b, P.qkvbuf, nullptr, Mt, 3 * CB, CB);
  attention_kernel<<<dim3(BNB, NHH), b128, 0, s>>>(P.qkvbuf, P.otok, k + 1, Npad);
  gemm_bf16_wmma<0><<<dim3(Mt / BM, CB / BN), b256, 0, s>>>(
      P.otok, P.po_wh, P.po_b, P.oproj, nullptr, Mt, CB, CB);

  bnpack_kernel<<<dim3((u32)(MC / 256)), b256, 0, s>>>(P.tLn, P.bn_g, P.bn_b, P.bn_m, P.bn_v, P.h1, MC, CB);
  gemm_bf16_wmma<1><<<dim3(M / BM, CB / BN), b256, 0, s>>>(
      P.h1, P.fc1_wh, P.fc1_b, nullptr, P.h2b, M, CB, CB);
  dwconv_tm_kernel<<<dim3((u32)(MC / 256)), b256, 0, s>>>(P.h2b, P.conv_w, P.conv_b, P.h1, CB, H, W, 0);
  gemm_bf16_wmma<0><<<dim3(M / BM, CB / BN), b256, 0, s>>>(
      P.h1, P.fc2_wh, P.fc2_b, P.tY3, nullptr, M, CB, CB);

  combine_kernel<<<dim3((u32)(MC / 256)), b256, 0, s>>>(P.tY3, P.oproj, P.tComb, Np, k, Npad, MC);
  scatter_kernel<<<dim3((u32)(((size_t)BNB * k * CB + 255) / 256)), b256, 0, s>>>(
      P.tY3, P.oproj, P.idx, P.tComb, Np, k, Npad);
  pack_bf16_kernel<<<dim3((u32)(MC / 256)), b256, 0, s>>>(P.tComb, P.h1, MC);
  gemm_bf16_wmma<0><<<dim3(M / BM, CB / BN), b256, 0, s>>>(
      P.h1, P.proj_wh, P.proj_b, P.tLn, nullptr, M, CB, CB);   // salout reuses tLn
  add_kernel<<<dim3((u32)(MC / 256)), b256, 0, s>>>(P.tA, P.tLn, P.tT2, MC);

  ln_rows<false, true, false><<<dim3((u32)M), b256, 0, s>>>(P.tT2, ln2g, ln2b, nullptr, P.h1, nullptr);
  gemm_bf16_wmma<1><<<dim3(M / BM, FDC / BN), b256, 0, s>>>(
      P.h1, P.f1_wh, P.f1_b, nullptr, P.hF1, M, FDC, CB);
  dwconv_tm_kernel<<<dim3((u32)(MF / 256)), b256, 0, s>>>(P.hF1, P.fdw_w, P.fdw_b, P.hF2, FDC, H, W, 1);
  gemm_bf16_wmma<0><<<dim3(M / BM, CB / BN), b256, 0, s>>>(
      P.hF2, P.f2_wh, P.f2_b, P.tY3, nullptr, M, CB, FDC);     // ffnout reuses tY3
  final_kernel<<<dim3((u32)(MC / 256)), b256, 0, s>>>(P.tT2, P.tY3, outp, Np, MC);
}

extern "C" void kernel_launch(void* const* d_in, const int* in_sizes, int n_in,
                              void* d_out, int out_size, void* d_ws, size_t ws_size,
                              hipStream_t stream)
{
  (void)in_sizes; (void)n_in; (void)out_size; (void)ws_size;

  size_t off = 0;
  auto alloc = [&](size_t bytes) -> void* {
    off = (off + 255) & ~(size_t)255;
    void* p = (char*)d_ws + off;
    off += bytes;
    return p;
  };

  const size_t Mx = 32768;            // x tokens (z fits inside)
  const size_t MCx = Mx * CB, MFx = Mx * FDC, Mtx = (size_t)BNB * 112;

  Params P;
  P.tA    = (float*)alloc(MCx * 4);
  P.tLn   = (float*)alloc(MCx * 4);
  P.tY3   = (float*)alloc(MCx * 4);
  P.tComb = (float*)alloc(MCx * 4);
  P.tT2   = (float*)alloc(MCx * 4);
  P.h1    = (u16*)alloc(MCx * 2);
  P.h2b   = (u16*)alloc(MCx * 2);
  P.hF1   = (u16*)alloc(MFx * 2);
  P.hF2   = (u16*)alloc(MFx * 2);
  P.tok   = (u16*)alloc(Mtx * CB * 2);
  P.otok  = (u16*)alloc(Mtx * CB * 2);
  P.qkvbuf = (float*)alloc(Mtx * 3 * CB * 4);
  P.oproj  = (float*)alloc(Mtx * CB * 4);
  P.scores = (float*)alloc(Mx * 4);
  P.chansum = (float*)alloc((size_t)BNB * CB * 4);
  P.idx     = (int*)alloc((size_t)BNB * 98 * 4);
  u16* qkv_wh  = (u16*)alloc((size_t)768 * 256 * 2);
  u16* po_wh   = (u16*)alloc((size_t)256 * 256 * 2);
  u16* fc1_wh  = (u16*)alloc((size_t)256 * 256 * 2);
  u16* fc2_wh  = (u16*)alloc((size_t)256 * 256 * 2);
  u16* proj_wh = (u16*)alloc((size_t)256 * 256 * 2);
  u16* f1_wh   = (u16*)alloc((size_t)1024 * 256 * 2);
  u16* f2_wh   = (u16*)alloc((size_t)256 * 1024 * 2);

  // pack all GEMM weights fp32 -> bf16
  pack_bf16_kernel<<<768 * 256 / 256, 256, 0, stream>>>((const float*)d_in[8],  qkv_wh,  (size_t)768 * 256);
  pack_bf16_kernel<<<256 * 256 / 256, 256, 0, stream>>>((const float*)d_in[10], po_wh,   (size_t)256 * 256);
  pack_bf16_kernel<<<256 * 256 / 256, 256, 0, stream>>>((const float*)d_in[16], fc1_wh,  (size_t)256 * 256);
  pack_bf16_kernel<<<256 * 256 / 256, 256, 0, stream>>>((const float*)d_in[20], fc2_wh,  (size_t)256 * 256);
  pack_bf16_kernel<<<256 * 256 / 256, 256, 0, stream>>>((const float*)d_in[22], proj_wh, (size_t)256 * 256);
  pack_bf16_kernel<<<1024 * 256 / 256, 256, 0, stream>>>((const float*)d_in[28], f1_wh,  (size_t)1024 * 256);
  pack_bf16_kernel<<<256 * 1024 / 256, 256, 0, stream>>>((const float*)d_in[32], f2_wh,  (size_t)256 * 1024);

  P.pos_w = (const float*)d_in[2];  P.pos_b = (const float*)d_in[3];
  P.qkv_b = (const float*)d_in[9];  P.po_b  = (const float*)d_in[11];
  P.bn_g  = (const float*)d_in[12]; P.bn_b  = (const float*)d_in[13];
  P.bn_m  = (const float*)d_in[14]; P.bn_v  = (const float*)d_in[15];
  P.fc1_b = (const float*)d_in[17];
  P.conv_w = (const float*)d_in[18]; P.conv_b = (const float*)d_in[19];
  P.fc2_b = (const float*)d_in[21]; P.proj_b = (const float*)d_in[23];
  P.f1_b  = (const float*)d_in[29];
  P.fdw_w = (const float*)d_in[30]; P.fdw_b = (const float*)d_in[31];
  P.f2_b  = (const float*)d_in[33];
  P.qkv_wh = qkv_wh; P.po_wh = po_wh; P.fc1_wh = fc1_wh; P.fc2_wh = fc2_wh;
  P.proj_wh = proj_wh; P.f1_wh = f1_wh; P.f2_wh = f2_wh;

  float* outZ = (float*)d_out;
  float* outX = outZ + (size_t)BNB * CB * 16 * 16;

  // z: 16x16, k=32 -> N=33 pad 48 ; x: 32x32, k=98 -> N=99 pad 112
  run_tensor((const float*)d_in[0], outZ, 16, 16, 32, 48,
             (const float*)d_in[4], (const float*)d_in[5],
             (const float*)d_in[24], (const float*)d_in[25], P, stream);
  run_tensor((const float*)d_in[1], outX, 32, 32, 98, 112,
             (const float*)d_in[6], (const float*)d_in[7],
             (const float*)d_in[26], (const float*)d_in[27], P, stream);
}
